// GNN_936302870769
// MI455X (gfx1250) — matrix operand order repaired and checked by reading, not compiled, not run on hardware
//
#include <hip/hip_runtime.h>
#include <hip/hip_bf16.h>
#include <math.h>

// ---------------------------------------------------------------------------
// GCN pipeline for MI455X (gfx1250, wave32).
//  - Node GEMMs via v_wmma_f32_16x16x32_bf16 (bf16 in, f32 accumulate).
//  - W staged into LDS by the Tensor Data Mover (tensor_load_to_lds) when the
//    builtin is available; one DMA per block instead of 16K VMEM loads.
//  - Edge gather/scatter is L2-resident (51MB working set << 192MB L2);
//    native global_atomic_add_f32 into agg.
//  - All non-GEMM math in fp32 to track the fp32 reference.
// ---------------------------------------------------------------------------

typedef __attribute__((ext_vector_type(16))) __bf16 v16bf;
typedef __attribute__((ext_vector_type(8)))  float  v8f;
typedef __attribute__((ext_vector_type(4)))  unsigned int v4u;
typedef __attribute__((ext_vector_type(8)))  int    v8i;
typedef __attribute__((ext_vector_type(4)))  int    v4i;

#define NN 100000
#define NE 1600000
#define DD 128
#define NG 64
#define DDENSE 512
#define NCLS 16

#if __has_builtin(__builtin_amdgcn_tensor_load_to_lds) && \
    __has_builtin(__builtin_amdgcn_s_wait_tensorcnt)
#define USE_TDM 1
#else
#define USE_TDM 0
#endif

// ---------------------------------------------------------------------------
// GEMM: H[M x 128] = X[M x 128] @ W[128 x 128], bf16 WMMA, f32 accumulate.
// 8 waves/block. W is DMA'd raw into LDS by the TDM (wave 0 issues one
// tensor_load_to_lds; TENSORcnt is per-wave, so wave 0 waits, then the block
// barrier releases everyone), then converted/swizzled to the CDNA5 B-operand
// layout in a second LDS region.
// ---------------------------------------------------------------------------
__global__ __launch_bounds__(256)
void k_gemm(const float* __restrict__ X, const float* __restrict__ W,
            float* __restrict__ H, int mtiles) {
    __shared__ alignas(16) float         ldsRawW[DD * DD];   // 64 KB raw f32 W
    __shared__ alignas(32) unsigned char ldsW[32 * 1024];    // 32 KB bf16 swizzled

#if USE_TDM
    if (threadIdx.x < 32) {   // wave 0 only: one TDM DMA per block
        unsigned long long ga = (unsigned long long)(uintptr_t)W;        // 57-bit
        unsigned int lds_off  = (unsigned int)(uintptr_t)ldsRawW;        // LDS aperture: low 32 bits = LDS byte offset
        // D# group 0 (ISA 08 §8.3): count=1, lds_addr, global_addr, type=2
        v4u g0 = { 1u,                       // count=1, is_restore=0, gather off
                   lds_off,                  // lds_addr [63:32]
                   (unsigned int)(ga & 0xffffffffu),            // global_addr lo
                   (unsigned int)((ga >> 32) & 0x01ffffffu) | (2u << 30) }; // addr hi + type=2
        // D# group 1 (§8.4): 2-D tensor, data_size=4B (code 2), no mask/pad/iterate
        // tensor_dim0 = 128, tensor_dim1 = 128, tile 128x128, dim0_stride = 128
        v8i g1 = { (int)(2u << 16),          // workgroup_mask=0 | data_size=2
                   (int)(128u << 16),        // barrier_addr=0 | tensor_dim0[15:0]=128
                   (int)(128u << 16),        // tensor_dim0 hi=0 | tensor_dim1[15:0]=128
                   (int)(128u << 16),        // tensor_dim1 hi=0 | tile_dim0=128
                   (int)(128u),              // tile_dim1=128 | tile_dim2=0
                   (int)(128u),              // tensor_dim0_stride[31:0]=128
                   0,                        // stride hi | tensor_dim1_stride lo (unused, 2-D)
                   0 };
        v4i gz = { 0, 0, 0, 0 };             // groups 2/3 unused (<=2-D tensor)
#if defined(__clang_major__) && __clang_major__ >= 23
        v8i gz8 = { 0, 0, 0, 0, 0, 0, 0, 0 };
        __builtin_amdgcn_tensor_load_to_lds(g0, g1, gz, gz, gz8, 0);
#else
        __builtin_amdgcn_tensor_load_to_lds(g0, g1, gz, gz, 0);
#endif
        __builtin_amdgcn_s_wait_tensorcnt(0);   // TENSORcnt is per-wave
    }
    __syncthreads();                            // release other waves
#else
    for (int idx = threadIdx.x; idx < DD * DD; idx += 256)
        ldsRawW[idx] = W[idx];
    __syncthreads();
#endif

    // Convert/swizzle W (f32 -> bf16) into WMMA B-operand layout:
// B 32x16 (KxN) per (kc,nt): lane = (n&15) | (((kr>>3)&1)<<4), element s = (kr&7) + ((kr>>4)<<3).
    for (int idx = threadIdx.x; idx < DD * DD; idx += 256) {
        int k = idx >> 7, n = idx & 127;
        int kc = k >> 5, kr = k & 31;
        int half = (kr >> 3) & 1, krr = (kr & 7) | ((kr >> 4) << 3);
        int v = krr >> 1, lo = krr & 1;
        int lane = (n & 15) | (half << 4);
        int nt = n >> 4;
        int off = ((kc * 8 + nt) * 32 + lane) * 32 + v * 4 + lo * 2;
        *(__bf16*)(ldsW + off) = (__bf16)ldsRawW[idx];
    }
    __syncthreads();

    int wave = threadIdx.x >> 5;
    int lane = threadIdx.x & 31;
    int tile = blockIdx.x * 8 + wave;
    if (tile >= mtiles) tile = mtiles - 1;   // keep EXEC all-1s (WMMA requirement);
                                             // extra waves duplicate identical stores.
    // A-operand (16x32 bf16, MxK) per lane: m = lane&15, khalf = (lane>>4)*8,
    // VGPR v<4 -> K=khalf+2v, v>=4 -> K=khalf+16+2(v-4).
    int mrow  = tile * 16 + (lane & 15);
    int khalf = (lane >> 4) * 8;
    v16bf a[4];
#pragma unroll
    for (int kc = 0; kc < 4; ++kc) {
#pragma unroll
        for (int v = 0; v < 8; ++v) {
            int k0 = kc * 32 + khalf + (v < 4 ? 2 * v : 16 + 2 * (v - 4));
            float2 f = *(const float2*)(X + (size_t)mrow * DD + k0);
            a[kc][2 * v]     = (__bf16)f.x;
            a[kc][2 * v + 1] = (__bf16)f.y;
        }
    }

#pragma unroll
    for (int nt = 0; nt < 8; ++nt) {
        v8f acc = {};
#pragma unroll
        for (int kc = 0; kc < 4; ++kc) {
            v16bf bmat = *(const v16bf*)(ldsW + (size_t)((kc * 8 + nt) * 32 + lane) * 32);
            acc = __builtin_amdgcn_wmma_f32_16x16x32_bf16(
                false, a[kc], false, bmat, (short)0, acc, false, false);
        }
        // C/D layout: VGPR v -> M = (lane>>4)*8 + v, N = nt*16 + (lane&15)
        int n    = nt * 16 + (lane & 15);
        int mtop = tile * 16 + (lane >> 4) * 8;
#pragma unroll
        for (int v = 0; v < 8; ++v)
            H[(size_t)(mtop + v) * DD + n] = acc[v];
    }
}

// ---------------------------------------------------------------------------
__global__ void k_zero(float* __restrict__ p, long n) {
    long i = (long)blockIdx.x * blockDim.x + threadIdx.x;
    if (i < n) p[i] = 0.0f;
}

__global__ void k_deg(const int* __restrict__ dst, float* __restrict__ deg, int ne) {
    int e = blockIdx.x * blockDim.x + threadIdx.x;
    if (e < ne)
        __hip_atomic_fetch_add(&deg[dst[e]], 1.0f, __ATOMIC_RELAXED, __HIP_MEMORY_SCOPE_AGENT);
}

__global__ void k_dinv(float* __restrict__ d, int n) {
    int i = blockIdx.x * blockDim.x + threadIdx.x;
    if (i < n) d[i] = rsqrtf(d[i] + 1.0f);   // +1 self loop
}

// One wave per edge: lanes cover the 128-float row as float4 each.
__global__ __launch_bounds__(256)
void k_scatter(const float* __restrict__ H, const float* __restrict__ dinv,
               const int* __restrict__ src, const int* __restrict__ dst,
               float* __restrict__ agg) {
    int e = blockIdx.x * 8 + (threadIdx.x >> 5);
    if (e >= NE) return;
    int lane = threadIdx.x & 31;
    int s = src[e], d = dst[e];
    if (e + 8 < NE)   // prefetch the next edge's source row (global_prefetch_b8)
        __builtin_prefetch(H + (size_t)src[e + 8] * DD, 0, 1);
    float c = dinv[s] * dinv[d];
    float4 hv = ((const float4*)(H + (size_t)s * DD))[lane];
    float* o = agg + (size_t)d * DD + lane * 4;
    __hip_atomic_fetch_add(o + 0, hv.x * c, __ATOMIC_RELAXED, __HIP_MEMORY_SCOPE_AGENT);
    __hip_atomic_fetch_add(o + 1, hv.y * c, __ATOMIC_RELAXED, __HIP_MEMORY_SCOPE_AGENT);
    __hip_atomic_fetch_add(o + 2, hv.z * c, __ATOMIC_RELAXED, __HIP_MEMORY_SCOPE_AGENT);
    __hip_atomic_fetch_add(o + 3, hv.w * c, __ATOMIC_RELAXED, __HIP_MEMORY_SCOPE_AGENT);
}

// out = agg + h * dinv^2 + bias   (self-loop contribution + bias)
__global__ void k_fin(const float* __restrict__ agg, const float* __restrict__ H,
                      const float* __restrict__ dinv, const float* __restrict__ bias,
                      float* __restrict__ out) {
    long i = (long)blockIdx.x * blockDim.x + threadIdx.x;
    if (i >= (long)NN * DD) return;
    int node = (int)(i >> 7), c = (int)(i & 127);
    float di = dinv[node];
    out[i] = agg[i] + H[i] * di * di + bias[c];
}

// ReLU + LayerNorm, one wave32 per node, butterfly reduction.
__global__ __launch_bounds__(256)
void k_relu_ln(const float* __restrict__ in, const float* __restrict__ g,
               const float* __restrict__ b, float* __restrict__ out) {
    int node = blockIdx.x * 8 + (threadIdx.x >> 5);
    if (node >= NN) return;
    int lane = threadIdx.x & 31;
    float4 v = ((const float4*)(in + (size_t)node * DD))[lane];
    v.x = fmaxf(v.x, 0.0f); v.y = fmaxf(v.y, 0.0f);
    v.z = fmaxf(v.z, 0.0f); v.w = fmaxf(v.w, 0.0f);
    float s  = v.x + v.y + v.z + v.w;
    float sq = v.x * v.x + v.y * v.y + v.z * v.z + v.w * v.w;
#pragma unroll
    for (int m = 16; m >= 1; m >>= 1) {
        s  += __shfl_xor(s,  m, 32);
        sq += __shfl_xor(sq, m, 32);
    }
    float mean = s * (1.0f / 128.0f);
    float var  = sq * (1.0f / 128.0f) - mean * mean;
    float r = rsqrtf(var + 1e-5f);
    float4 gv = ((const float4*)g)[lane];
    float4 bv = ((const float4*)b)[lane];
    float4 o;
    o.x = (v.x - mean) * r * gv.x + bv.x;
    o.y = (v.y - mean) * r * gv.y + bv.y;
    o.z = (v.z - mean) * r * gv.z + bv.z;
    o.w = (v.w - mean) * r * gv.w + bv.w;
    ((float4*)(out + (size_t)node * DD))[lane] = o;
}

// Global mean-pool accumulate: one wave per node.
__global__ __launch_bounds__(256)
void k_pool(const float* __restrict__ X, const int* __restrict__ batch,
            float* __restrict__ pooled, float* __restrict__ cnt) {
    int node = blockIdx.x * 8 + (threadIdx.x >> 5);
    if (node >= NN) return;
    int lane = threadIdx.x & 31;
    int gid = batch[node];
    float4 v = ((const float4*)(X + (size_t)node * DD))[lane];
    float* o = pooled + (size_t)gid * DD + lane * 4;
    __hip_atomic_fetch_add(o + 0, v.x, __ATOMIC_RELAXED, __HIP_MEMORY_SCOPE_AGENT);
    __hip_atomic_fetch_add(o + 1, v.y, __ATOMIC_RELAXED, __HIP_MEMORY_SCOPE_AGENT);
    __hip_atomic_fetch_add(o + 2, v.z, __ATOMIC_RELAXED, __HIP_MEMORY_SCOPE_AGENT);
    __hip_atomic_fetch_add(o + 3, v.w, __ATOMIC_RELAXED, __HIP_MEMORY_SCOPE_AGENT);
    if (lane == 0)
        __hip_atomic_fetch_add(&cnt[gid], 1.0f, __ATOMIC_RELAXED, __HIP_MEMORY_SCOPE_AGENT);
}

// Tiny MLP + log_softmax: one block per graph.
__global__ __launch_bounds__(256)
void k_mlp(const float* __restrict__ pooled, const float* __restrict__ cnt,
           const float* __restrict__ W1, const float* __restrict__ b1,
           const float* __restrict__ W2, const float* __restrict__ b2,
           const float* __restrict__ W3, const float* __restrict__ b3,
           float* __restrict__ out) {
    __shared__ float p[DD], z1[DDENSE], z2[DDENSE / 2], z3[NCLS], red[2];
    int g = blockIdx.x, t = threadIdx.x;
    if (t < DD) p[t] = pooled[(size_t)g * DD + t] / fmaxf(cnt[g], 1.0f);
    __syncthreads();
    for (int j = t; j < DDENSE; j += 256) {
        float s = b1[j];
        for (int k = 0; k < DD; ++k) s += p[k] * W1[k * DDENSE + j];
        z1[j] = s;
    }
    __syncthreads();
    {
        float s = b2[t & 255];
        int j = t;  // t < 256 always
        for (int k = 0; k < DDENSE; ++k) s += z1[k] * W2[k * (DDENSE / 2) + j];
        z2[j] = s;
    }
    __syncthreads();
    if (t < NCLS) {
        float s = b3[t];
        for (int k = 0; k < DDENSE / 2; ++k) s += z2[k] * W3[k * NCLS + t];
        z3[t] = s;
    }
    __syncthreads();
    if (t == 0) {
        float m = z3[0];
        for (int c = 1; c < NCLS; ++c) m = fmaxf(m, z3[c]);
        float sum = 0.0f;
        for (int c = 0; c < NCLS; ++c) sum += expf(z3[c] - m);
        red[0] = m; red[1] = logf(sum);
    }
    __syncthreads();
    if (t < NCLS) out[(size_t)g * NCLS + t] = z3[t] - red[0] - red[1];
}

// ---------------------------------------------------------------------------
extern "C" void kernel_launch(void* const* d_in, const int* in_sizes, int n_in,
                              void* d_out, int out_size, void* d_ws, size_t ws_size,
                              hipStream_t stream) {
    const float* x     = (const float*)d_in[0];
    const int*   ei    = (const int*)d_in[1];     // [2, NE]: row0 = src, row1 = dst
    const int*   srcv  = ei;
    const int*   dstv  = ei + NE;
    const int*   batch = (const int*)d_in[3];
    const float* W0    = (const float*)d_in[4];
    const float* b0    = (const float*)d_in[5];
    const float* convW = (const float*)d_in[6];   // [3,128,128]
    const float* convb = (const float*)d_in[7];   // [3,128]
    const float* lng   = (const float*)d_in[8];   // [4,128]
    const float* lnb   = (const float*)d_in[9];
    const float* W1    = (const float*)d_in[10];
    const float* b1    = (const float*)d_in[11];
    const float* W2    = (const float*)d_in[12];
    const float* b2    = (const float*)d_in[13];
    const float* W3    = (const float*)d_in[14];
    const float* b3    = (const float*)d_in[15];

    float* out      = (float*)d_out;
    float* emb_out  = out;                         // [NN, 128]
    float* logp_out = out + (size_t)NN * DD;       // [64, 16]

    // Workspace layout (floats). Needs ~155 MB.
    float* ws = (float*)d_ws;
    size_t o = 0;
    float* dinv   = ws + o; o += 100352;               // deg, then d^-1/2
    float* hb     = ws + o; o += (size_t)NN * DD;      // h = x @ W
    float* agg    = ws + o; o += (size_t)NN * DD;      // neighbor aggregate
    float* xbuf   = ws + o; o += (size_t)NN * DD;      // relu+LN output
    float* pooled = ws + o; o += (size_t)NG * DD;      // pooled sums
    float* cnt    = ws + o; o += 64;                   // per-graph counts (contiguous after pooled)

    const long nd = (long)NN * DD;
    const int mtiles = NN / 16;                        // 6250

    // degree -> dinv
    k_zero<<<(NN + 255) / 256, 256, 0, stream>>>(dinv, NN);
    k_deg <<<(NE + 255) / 256, 256, 0, stream>>>(dstv, dinv, NE);
    k_dinv<<<(NN + 255) / 256, 256, 0, stream>>>(dinv, NN);

    for (int l = 0; l < 4; ++l) {
        const float* xin = (l == 0) ? x : xbuf;
        const float* Wl  = (l == 0) ? W0 : convW + (size_t)(l - 1) * DD * DD;
        const float* bl  = (l == 0) ? b0 : convb + (size_t)(l - 1) * DD;
        k_gemm<<<(mtiles + 7) / 8, 256, 0, stream>>>(xin, Wl, hb, mtiles);
        k_zero<<<(int)((nd + 255) / 256), 256, 0, stream>>>(agg, nd);
        k_scatter<<<(NE + 7) / 8, 256, 0, stream>>>(hb, dinv, srcv, dstv, agg);
        float* conv = (l == 3) ? emb_out : agg;        // last conv output == emb
        k_fin<<<(int)((nd + 255) / 256), 256, 0, stream>>>(agg, hb, dinv, bl, conv);
        k_relu_ln<<<(NN + 7) / 8, 256, 0, stream>>>(conv, lng + (size_t)l * DD,
                                                    lnb + (size_t)l * DD, xbuf);
    }

    // pooled + cnt are contiguous: zero both in one shot
    k_zero<<<(NG * DD + 64 + 255) / 256, 256, 0, stream>>>(pooled, NG * DD + 64);
    k_pool<<<(NN + 7) / 8, 256, 0, stream>>>(xbuf, batch, pooled, cnt);
    k_mlp <<<NG, 256, 0, stream>>>(pooled, cnt, W1, b1, W2, b2, W3, b3, logp_out);
}
